// Room_1623497637931
// MI455X (gfx1250) — compile-verified, matrix operand order checked by reading, not run on hardware
//
#include <hip/hip_runtime.h>
#include <hip/hip_bf16.h>
#include <math.h>

// ---------------------------------------------------------------------------
// Transformer block for MI455X (gfx1250, wave32, WMMA).
// GEMMs: v_wmma_f32_16x16x32_bf16 with double-buffered LDS staging fed by
// async global->LDS copies (ASYNCcnt) when the toolchain exposes them.
// ---------------------------------------------------------------------------

#define D_MODEL 1024
#define N_HEADS 16
#define D_HEAD  64
#define D_FF    4096
#define SEQ_T   2048
#define NBATCH  4
#define BT      (NBATCH * SEQ_T)   // 8192 rows

#define LDS_LD  40                 // bf16 elements per staged row (80B pitch)

typedef __attribute__((ext_vector_type(16))) __bf16       v16bf;
typedef __attribute__((ext_vector_type(8)))  float        v8f;
typedef __attribute__((ext_vector_type(4)))  unsigned int u32x4;
typedef __attribute__((ext_vector_type(4)))  int          v4i;
typedef __attribute__((ext_vector_type(4)))  float        f32x4;

#if defined(__gfx1250__) && __has_builtin(__builtin_amdgcn_global_load_async_to_lds_b128)
#define ASYNC_COPY 1
// address-space-qualified pointee types for the async-copy builtin
typedef __attribute__((address_space(1))) v4i as1_v4i;
typedef __attribute__((address_space(3))) v4i as3_v4i;
#else
#define ASYNC_COPY 0
#endif

union FragU {
  u32x4  q[2];
  __bf16 e[16];
  v16bf  v;
};

__device__ __forceinline__ void wait_async0() {
#if defined(__gfx1250__)
#if __has_builtin(__builtin_amdgcn_s_wait_asynccnt)
  __builtin_amdgcn_s_wait_asynccnt(0);
#else
  asm volatile("s_wait_asynccnt 0" ::: "memory");
#endif
#endif
}

// Copy one 16B chunk global -> LDS (async if available).
__device__ __forceinline__ void copy16(const __bf16* __restrict__ g, __bf16* l) {
#if ASYNC_COPY
  __builtin_amdgcn_global_load_async_to_lds_b128((as1_v4i*)g, (as3_v4i*)l, 0, 0);
#else
  *reinterpret_cast<u32x4*>(l) = *reinterpret_cast<const u32x4*>(g);
#endif
}

// CDNA5 WMMA 16x32 bf16 A/B fragment layout:
// lane l: row = row0 + (l&15); hb = l>>4; elems 0..7 -> K = k0+hb*8+j,
// elems 8..15 -> K = k0+16+hb*8+(j-8). Two 16-byte loads per lane.
__device__ __forceinline__ v16bf load_frag(const __bf16* __restrict__ base,
                                           int ld, int row0, int k0, int lane) {
  const int m  = row0 + (lane & 15);
  const int hb = (lane >> 4) & 1;
  const __bf16* p = base + (size_t)m * ld + k0 + hb * 8;
  FragU u;
  u.q[0] = *reinterpret_cast<const u32x4*>(p);
  u.q[1] = *reinterpret_cast<const u32x4*>(p + 16);
  return u.v;
}

// Fragment load from a staged LDS tile (row pitch LDS_LD, conflict-free).
__device__ __forceinline__ v16bf load_frag_lds(const __bf16* base, int row0, int lane) {
  const int m  = row0 + (lane & 15);
  const int hb = (lane >> 4) & 1;
  const __bf16* p = base + m * LDS_LD + hb * 8;
  FragU u;
  u.q[0] = *reinterpret_cast<const u32x4*>(p);
  u.q[1] = *reinterpret_cast<const u32x4*>(p + 16);
  return u.v;
}

__device__ __forceinline__ v8f wmma_bf16(v16bf a, v16bf b, v8f c) {
  return __builtin_amdgcn_wmma_f32_16x16x32_bf16(false, a, false, b,
                                                 (short)0, c, false, false);
}

// ---------------------------------------------------------------------------
// Weight convert + transpose: src f32 [K][N] -> dst bf16 [N][K]
// ---------------------------------------------------------------------------
__global__ __launch_bounds__(256) void transpose_to_bf16(
    const float* __restrict__ src, __bf16* __restrict__ dst, int K, int N) {
  __shared__ float tile[32][33];
  const int kb = blockIdx.x * 32;
  const int nb = blockIdx.y * 32;
  const int tx = threadIdx.x;          // 0..31
  const int ty = threadIdx.y;          // 0..7
  #pragma unroll
  for (int i = ty; i < 32; i += 8)
    tile[i][tx] = src[(size_t)(kb + i) * N + nb + tx];
  __syncthreads();
  #pragma unroll
  for (int i = ty; i < 32; i += 8)
    dst[(size_t)(nb + i) * K + kb + tx] = (__bf16)tile[tx][i];
}

// ---------------------------------------------------------------------------
// LayerNorm over rows of 1024, write bf16
// ---------------------------------------------------------------------------
__global__ __launch_bounds__(256) void ln_bf16_kernel(
    const float* __restrict__ x, const float* __restrict__ gamma,
    const float* __restrict__ beta, __bf16* __restrict__ out) {
  __shared__ float rs1[256];
  __shared__ float rs2[256];
  const int row = blockIdx.x;
  const int tid = threadIdx.x;
  const float* xr = x + (size_t)row * D_MODEL;
  f32x4 v = *reinterpret_cast<const f32x4*>(xr + tid * 4);
  rs1[tid] = v[0] + v[1] + v[2] + v[3];
  rs2[tid] = v[0]*v[0] + v[1]*v[1] + v[2]*v[2] + v[3]*v[3];
  __syncthreads();
  for (int off = 128; off > 0; off >>= 1) {
    if (tid < off) { rs1[tid] += rs1[tid + off]; rs2[tid] += rs2[tid + off]; }
    __syncthreads();
  }
  const float mu   = rs1[0] * (1.0f / D_MODEL);
  const float var  = rs2[0] * (1.0f / D_MODEL) - mu * mu;
  const float rstd = rsqrtf(var + 1e-5f);
  #pragma unroll
  for (int i = 0; i < 4; i++) {
    const int c = tid * 4 + i;
    out[(size_t)row * D_MODEL + c] =
        (__bf16)((v[i] - mu) * rstd * gamma[c] + beta[c]);
  }
}

// ---------------------------------------------------------------------------
// Stage one 128x32 bf16 tile of A and of Bt into LDS (2 x 16B chunks per
// thread per tile; 256 threads cover 128 rows x 4 chunks).
// ---------------------------------------------------------------------------
__device__ __forceinline__ void stage_tiles(
    const __bf16* __restrict__ gA, const __bf16* __restrict__ gB, int K,
    __bf16* lA, __bf16* lB, int k0, int tid) {
  const int c0  = tid * 2;
  const int row = c0 >> 2;   // 0..127
  const int sub = c0 & 3;    // 0 or 2
  const __bf16* ga = gA + (size_t)row * K + k0 + sub * 8;
  const __bf16* gb = gB + (size_t)row * K + k0 + sub * 8;
  __bf16* la = lA + row * LDS_LD + sub * 8;
  __bf16* lb = lB + row * LDS_LD + sub * 8;
  copy16(ga,     la);
  copy16(ga + 8, la + 8);
  copy16(gb,     lb);
  copy16(gb + 8, lb + 8);
}

// ---------------------------------------------------------------------------
// Generic bf16 WMMA GEMM: C[M,N] = A[M,K] * Bt[N,K]^T (+bias, fused epilogue)
// 256 threads = 8 waves; macro-tile 128x128; each wave 32x64 (2x4 WMMA tiles).
// Double-buffered LDS staging with async global->LDS fills.
// MODE 0: out bf16 Q/K head layout [B,H,T,Dh]
// MODE 1: out bf16 V transposed layout [B,H,Dh,T]
// MODE 2: outF = res + (acc + bias)                    (x1 = x + attn_out)
// MODE 3: outB = bf16(gelu(acc + bias))                (FFN up)
// MODE 4: outF = x0 + gate[b]*(res + acc + bias - x0)  (final)
// ---------------------------------------------------------------------------
template <int MODE>
__global__ __launch_bounds__(256) void gemm_bf16(
    const __bf16* __restrict__ A, const __bf16* __restrict__ Bt,
    const float* __restrict__ bias,
    float* __restrict__ outF, __bf16* __restrict__ outB,
    const float* __restrict__ res, const float* __restrict__ x0,
    const float* __restrict__ gate,
    int M, int N, int K) {
  __shared__ __bf16 lsA[2][128 * LDS_LD];
  __shared__ __bf16 lsB[2][128 * LDS_LD];

  const int tid  = threadIdx.x;
  const int lane = tid & 31;
  const int wave = tid >> 5;
  const int wm = wave & 3;
  const int wn = wave >> 2;
  const int mBase = blockIdx.x * 128 + wm * 32;
  const int nBase = blockIdx.y * 128 + wn * 64;

  const __bf16* gA = A  + (size_t)(blockIdx.x * 128) * K;
  const __bf16* gB = Bt + (size_t)(blockIdx.y * 128) * K;

  v8f acc[2][4] = {};

  const int nk = K / 32;
  stage_tiles(gA, gB, K, lsA[0], lsB[0], 0, tid);

  for (int kt = 0; kt < nk; ++kt) {
    const int cur = kt & 1;
    wait_async0();
    __syncthreads();                 // buf[cur] resident for all waves
    if (kt + 1 < nk)                 // fill buf[cur^1] while computing
      stage_tiles(gA, gB, K, lsA[cur ^ 1], lsB[cur ^ 1], (kt + 1) * 32, tid);

    v16bf af[2], bfr[4];
    #pragma unroll
    for (int i = 0; i < 2; i++) af[i]  = load_frag_lds(lsA[cur], wm * 32 + 16 * i, lane);
    #pragma unroll
    for (int j = 0; j < 4; j++) bfr[j] = load_frag_lds(lsB[cur], wn * 64 + 16 * j, lane);
    #pragma unroll
    for (int i = 0; i < 2; i++)
      #pragma unroll
      for (int j = 0; j < 4; j++)
        acc[i][j] = wmma_bf16(af[i], bfr[j], acc[i][j]);
  }

  const int colL = lane & 15;
  const int hb   = (lane >> 4) & 1;
  #pragma unroll
  for (int i = 0; i < 2; i++) {
    #pragma unroll
    for (int j = 0; j < 4; j++) {
      const int nIdx = nBase + j * 16 + colL;
      const float bv = bias[nIdx];
      #pragma unroll
      for (int v = 0; v < 8; v++) {
        const int mIdx = mBase + i * 16 + v + 8 * hb;
        float val = acc[i][j][v] + bv;
        if constexpr (MODE == 0) {
          const int b = mIdx >> 11, t = mIdx & (SEQ_T - 1);
          const int h = nIdx >> 6, d = nIdx & (D_HEAD - 1);
          outB[(((size_t)(b * N_HEADS + h) * SEQ_T + t) * D_HEAD) + d] = (__bf16)val;
        } else if constexpr (MODE == 1) {
          const int b = mIdx >> 11, t = mIdx & (SEQ_T - 1);
          const int h = nIdx >> 6, d = nIdx & (D_HEAD - 1);
          outB[(((size_t)(b * N_HEADS + h) * D_HEAD + d) * SEQ_T) + t] = (__bf16)val;
        } else if constexpr (MODE == 2) {
          const size_t idx = (size_t)mIdx * N + nIdx;
          outF[idx] = res[idx] + val;
        } else if constexpr (MODE == 3) {
          const float ge = 0.5f * val * (1.0f + erff(val * 0.70710678118654752f));
          outB[(size_t)mIdx * N + nIdx] = (__bf16)ge;
        } else {  // MODE == 4
          const size_t idx = (size_t)mIdx * N + nIdx;
          const float xv  = x0[idx];
          const float x1v = res[idx];
          const float g   = gate[mIdx >> 11];
          outF[idx] = xv + g * (x1v + val - xv);
        }
      }
    }
  }
}

// ---------------------------------------------------------------------------
// Flash attention: one wave per 16-row q-tile, online softmax, all-WMMA.
// Q,K: bf16 [B,H,T,Dh]; Vt: bf16 [B,H,Dh,T]; ctx: bf16 [B*T, D_MODEL]
// grid = (T/128, B*H), block = 256 (8 waves, each a distinct q-tile).
// ---------------------------------------------------------------------------
__global__ __launch_bounds__(256) void attn_kernel(
    const __bf16* __restrict__ Q, const __bf16* __restrict__ Km,
    const __bf16* __restrict__ Vt, __bf16* __restrict__ ctx) {
  __shared__ float pbuf[8 * 16 * 33];   // per-wave 16x32 P staging (+pad)
  const int lane = threadIdx.x & 31;
  const int wave = threadIdx.x >> 5;
  const int bh = blockIdx.y;
  const int b  = bh >> 4;
  const int h  = bh & (N_HEADS - 1);
  const int q0 = blockIdx.x * 128 + wave * 16;

  const __bf16* Qb = Q  + (size_t)bh * SEQ_T * D_HEAD;
  const __bf16* Kb = Km + (size_t)bh * SEQ_T * D_HEAD;
  const __bf16* Vb = Vt + (size_t)bh * D_HEAD * SEQ_T;

  const v16bf qf0 = load_frag(Qb, D_HEAD, q0, 0,  lane);
  const v16bf qf1 = load_frag(Qb, D_HEAD, q0, 32, lane);

  v8f o[4] = {};
  float mrow[8], lrow[8];
  #pragma unroll
  for (int v = 0; v < 8; v++) { mrow[v] = -1e30f; lrow[v] = 0.0f; }

  const int colL = lane & 15;
  const int hb   = (lane >> 4) & 1;
  float* myp = &pbuf[wave * 16 * 33];

  for (int j0 = 0; j0 < q0 + 16; j0 += 32) {
    // S = Q K^T for 16 q-rows x 32 kv-cols (two 16x16 tiles)
    v8f s0 = {}, s1 = {};
    s0 = wmma_bf16(qf0, load_frag(Kb, D_HEAD, j0,      0,  lane), s0);
    s0 = wmma_bf16(qf1, load_frag(Kb, D_HEAD, j0,      32, lane), s0);
    s1 = wmma_bf16(qf0, load_frag(Kb, D_HEAD, j0 + 16, 0,  lane), s1);
    s1 = wmma_bf16(qf1, load_frag(Kb, D_HEAD, j0 + 16, 32, lane), s1);

    float alphav[8];
    #pragma unroll
    for (int v = 0; v < 8; v++) {
      const int row = q0 + v + 8 * hb;
      const int c0 = j0 + colL;
      const int c1 = j0 + 16 + colL;
      float a0 = s0[v] * 0.125f;  if (c0 > row) a0 = -1e30f;
      float a1 = s1[v] * 0.125f;  if (c1 > row) a1 = -1e30f;
      // row-max across the 16 lanes holding this row
      float rm = fmaxf(a0, a1);
      rm = fmaxf(rm, __shfl_xor(rm, 1, 32));
      rm = fmaxf(rm, __shfl_xor(rm, 2, 32));
      rm = fmaxf(rm, __shfl_xor(rm, 4, 32));
      rm = fmaxf(rm, __shfl_xor(rm, 8, 32));
      const float mnew  = fmaxf(mrow[v], rm);
      const float alpha = __expf(mrow[v] - mnew);
      const float p0 = __expf(a0 - mnew);
      const float p1 = __expf(a1 - mnew);
      float rs = p0 + p1;
      rs += __shfl_xor(rs, 1, 32);
      rs += __shfl_xor(rs, 2, 32);
      rs += __shfl_xor(rs, 4, 32);
      rs += __shfl_xor(rs, 8, 32);
      lrow[v] = lrow[v] * alpha + rs;
      mrow[v] = mnew;
      alphav[v] = alpha;
      const int r = v + 8 * hb;
      myp[r * 33 + colL]      = p0;
      myp[r * 33 + 16 + colL] = p1;
    }
    // rescale running O
    #pragma unroll
    for (int n = 0; n < 4; n++)
      #pragma unroll
      for (int v = 0; v < 8; v++)
        o[n][v] *= alphav[v];

    asm volatile("" ::: "memory");  // keep LDS store->load order (in-wave, in-order DS pipe)

    // reload P in A-fragment layout, convert to bf16
    FragU pf;
    const int m = lane & 15;
    #pragma unroll
    for (int j = 0; j < 8; j++) {
      pf.e[j]     = (__bf16)myp[m * 33 + hb * 8 + j];
      pf.e[8 + j] = (__bf16)myp[m * 33 + 16 + hb * 8 + j];
    }
    // O += P * V   (V^T rows are d, contiguous in t)
    #pragma unroll
    for (int n = 0; n < 4; n++) {
      const v16bf vf = load_frag(Vb, SEQ_T, n * 16, j0, lane);
      o[n] = wmma_bf16(pf.v, vf, o[n]);
    }
  }

  // normalize and write ctx in [B*T, D_MODEL] bf16 (column = h*64 + d)
  #pragma unroll
  for (int n = 0; n < 4; n++) {
    #pragma unroll
    for (int v = 0; v < 8; v++) {
      const int t = q0 + v + 8 * hb;
      const float val = o[n][v] * (1.0f / lrow[v]);
      ctx[((size_t)(b * SEQ_T + t) * D_MODEL) + h * D_HEAD + n * 16 + colL] = (__bf16)val;
    }
  }
}

// ---------------------------------------------------------------------------
// Host-side launcher
// ---------------------------------------------------------------------------
extern "C" void kernel_launch(void* const* d_in, const int* in_sizes, int n_in,
                              void* d_out, int out_size, void* d_ws, size_t ws_size,
                              hipStream_t stream) {
  const float* x      = (const float*)d_in[0];
  const float* gate   = (const float*)d_in[1];
  const float* Wq     = (const float*)d_in[2];
  const float* bq     = (const float*)d_in[3];
  const float* Wk     = (const float*)d_in[4];
  const float* bk     = (const float*)d_in[5];
  const float* Wv     = (const float*)d_in[6];
  const float* bv     = (const float*)d_in[7];
  const float* Wo     = (const float*)d_in[8];
  const float* bo     = (const float*)d_in[9];
  const float* W1     = (const float*)d_in[10];
  const float* b1     = (const float*)d_in[11];
  const float* W2     = (const float*)d_in[12];
  const float* b2     = (const float*)d_in[13];
  const float* gamma1 = (const float*)d_in[14];
  const float* beta1  = (const float*)d_in[15];
  const float* gamma2 = (const float*)d_in[16];
  const float* beta2  = (const float*)d_in[17];
  float* out = (float*)d_out;

  // workspace carve (256B aligned)
  char* p = (char*)d_ws;
  auto carve = [&](size_t bytes) -> char* {
    char* r = p;
    p += (bytes + 255) & ~(size_t)255;
    return r;
  };
  __bf16* hbf  = (__bf16*)carve((size_t)BT * D_MODEL * 2);
  __bf16* WqT  = (__bf16*)carve((size_t)D_MODEL * D_MODEL * 2);
  __bf16* WkT  = (__bf16*)carve((size_t)D_MODEL * D_MODEL * 2);
  __bf16* WvT  = (__bf16*)carve((size_t)D_MODEL * D_MODEL * 2);
  __bf16* WoT  = (__bf16*)carve((size_t)D_MODEL * D_MODEL * 2);
  __bf16* W1T  = (__bf16*)carve((size_t)D_FF * D_MODEL * 2);
  __bf16* W2T  = (__bf16*)carve((size_t)D_MODEL * D_FF * 2);
  __bf16* Qb   = (__bf16*)carve((size_t)BT * D_MODEL * 2);
  __bf16* Kbuf = (__bf16*)carve((size_t)BT * D_MODEL * 2);
  __bf16* Vtb  = (__bf16*)carve((size_t)BT * D_MODEL * 2);
  __bf16* ctx  = (__bf16*)carve((size_t)BT * D_MODEL * 2);
  float*  x1   = (float*) carve((size_t)BT * D_MODEL * 4);
  __bf16* h2   = (__bf16*)carve((size_t)BT * D_MODEL * 2);
  __bf16* ff1  = (__bf16*)carve((size_t)BT * D_FF * 2);

  const dim3 tb(32, 8);
  // weight transpose+convert: src [K][N] -> dst [N][K]
  transpose_to_bf16<<<dim3(D_MODEL/32, D_MODEL/32), tb, 0, stream>>>(Wq, WqT, D_MODEL, D_MODEL);
  transpose_to_bf16<<<dim3(D_MODEL/32, D_MODEL/32), tb, 0, stream>>>(Wk, WkT, D_MODEL, D_MODEL);
  transpose_to_bf16<<<dim3(D_MODEL/32, D_MODEL/32), tb, 0, stream>>>(Wv, WvT, D_MODEL, D_MODEL);
  transpose_to_bf16<<<dim3(D_MODEL/32, D_MODEL/32), tb, 0, stream>>>(Wo, WoT, D_MODEL, D_MODEL);
  transpose_to_bf16<<<dim3(D_MODEL/32, D_FF/32),    tb, 0, stream>>>(W1, W1T, D_MODEL, D_FF);
  transpose_to_bf16<<<dim3(D_FF/32, D_MODEL/32),    tb, 0, stream>>>(W2, W2T, D_FF, D_MODEL);

  // h = LN1(x) -> bf16
  ln_bf16_kernel<<<BT, 256, 0, stream>>>(x, gamma1, beta1, hbf);

  // Q/K/V projections
  const dim3 gQKV(BT/128, D_MODEL/128);
  gemm_bf16<0><<<gQKV, 256, 0, stream>>>(hbf, WqT, bq, nullptr, Qb,   nullptr, nullptr, nullptr, BT, D_MODEL, D_MODEL);
  gemm_bf16<0><<<gQKV, 256, 0, stream>>>(hbf, WkT, bk, nullptr, Kbuf, nullptr, nullptr, nullptr, BT, D_MODEL, D_MODEL);
  gemm_bf16<1><<<gQKV, 256, 0, stream>>>(hbf, WvT, bv, nullptr, Vtb,  nullptr, nullptr, nullptr, BT, D_MODEL, D_MODEL);

  // causal flash attention
  attn_kernel<<<dim3(SEQ_T/128, NBATCH*N_HEADS), 256, 0, stream>>>(Qb, Kbuf, Vtb, ctx);

  // x1 = x + ctx @ Wo + bo
  gemm_bf16<2><<<gQKV, 256, 0, stream>>>(ctx, WoT, bo, x1, nullptr, x, nullptr, nullptr, BT, D_MODEL, D_MODEL);

  // h2 = LN2(x1)
  ln_bf16_kernel<<<BT, 256, 0, stream>>>(x1, gamma2, beta2, h2);

  // ff1 = gelu(h2 @ W1 + b1)
  gemm_bf16<3><<<dim3(BT/128, D_FF/128), 256, 0, stream>>>(h2, W1T, b1, nullptr, ff1, nullptr, nullptr, nullptr, BT, D_FF, D_MODEL);

  // out = x + g*(x1 + ff1@W2 + b2 - x)
  gemm_bf16<4><<<dim3(BT/128, D_MODEL/128), 256, 0, stream>>>(ff1, W2T, b2, out, nullptr, x1, x, gate, BT, D_MODEL, D_FF);
}
